// MultiHeadAttention_20109036880147
// MI455X (gfx1250) — matrix operand order repaired and checked by reading, not compiled, not run on hardware
//
#include <hip/hip_runtime.h>
#include <hip/hip_bf16.h>

typedef __attribute__((ext_vector_type(16))) __bf16        v16bf;
typedef __attribute__((ext_vector_type(8)))  float         v8f;
typedef __attribute__((ext_vector_type(8)))  unsigned int  v8u;
typedef __attribute__((ext_vector_type(4)))  unsigned int  u32x4;
typedef __attribute__((ext_vector_type(8)))  unsigned int  u32x8;

// Problem dims (fixed by the reference).
#define BATCH 2
#define SEQL  2048
#define EMB   2048
#define NH    16
#define HD    128

// ---------------------------------------------------------------- helpers ---
static __device__ __forceinline__ unsigned short f2bf(float f) {
  unsigned u = __builtin_bit_cast(unsigned, f);
  u += 0x7FFFu + ((u >> 16) & 1u);          // round-to-nearest-even
  return (unsigned short)(u >> 16);
}

// Assemble a v16bf fragment from two contiguous 16-byte chunks.
static __device__ __forceinline__ v16bf ld_frag(const unsigned short* p0,
                                                const unsigned short* p1) {
  v8u t;
  ((uint4*)&t)[0] = *(const uint4*)p0;
  ((uint4*)&t)[1] = *(const uint4*)p1;
  return __builtin_bit_cast(v16bf, t);
}

// ---------------------------------------------------------- TDM tile load ---
// Issue one Tensor-Data-Mover 2D tile load (global -> LDS) per ISA 08, D#
// groups packed per 8.3/8.4. All parameters are wave-uniform. data_size = 8B.
//   lds_addr : LDS byte address of tile start (low 32 bits of shared ptr)
//   gptr     : global byte address of tile start
//   g1w0     : group1 word0 (data_size | pad config)
//   tdim0/1  : tensor extents (units / rows) for OOB handling
//   tile0/1  : tile extents (units / rows)
//   stride0  : row stride in 8B units
static __device__ __forceinline__ void tdm_load_2d(unsigned lds_addr,
                                                   const void* gptr,
                                                   unsigned g1w0,
                                                   unsigned tdim0, unsigned tdim1,
                                                   unsigned tile0, unsigned tile1,
                                                   unsigned stride0) {
  unsigned long long ga = (unsigned long long)gptr;
  u32x4 g0;
  g0[0] = 1u;                                        // count=1, user desc
  g0[1] = lds_addr;                                  // lds_addr
  g0[2] = (unsigned)ga;                              // global_addr[31:0]
  g0[3] = ((unsigned)(ga >> 32) & 0x1FFFFFFu) | 0x80000000u;  // addr[56:32] | type=2
  u32x8 g1;
  g1[0] = g1w0;                                      // mask=0 | data_size | pad
  g1[1] = tdim0 << 16;                               // tensor_dim0[15:0]
  g1[2] = (tdim0 >> 16) | (tdim1 << 16);             // tensor_dim0 hi | dim1 lo
  g1[3] = (tdim1 >> 16) | (tile0 << 16);             // tensor_dim1 hi | tile_dim0
  g1[4] = tile1;                                     // tile_dim1 (tile_dim2 = 0)
  g1[5] = stride0;                                   // tensor_dim0_stride[31:0]
  g1[6] = 0;                                         // stride hi | dim1_stride lo
  g1[7] = 0;
  asm volatile("tensor_load_to_lds %0, %1" :: "s"(g0), "s"(g1) : "memory");
}

// group1 word0: data_size=8B (3<<16), pad_enable (1<<20),
// pad_interval code I (2^(I+1) dwords), pad_amount code 3 (4 dwords = 16B).
#define G1W0_PAD(I) ((3u << 16) | (1u << 20) | ((unsigned)(I) << 22) | (3u << 25))

// ------------------------------------------------------------ cast kernel ---
__global__ void cast_f32_bf16(const float* __restrict__ in,
                              unsigned short* __restrict__ out, int n) {
  int i = blockIdx.x * blockDim.x + threadIdx.x;
  int stride = gridDim.x * blockDim.x;
  for (; i < n; i += stride) out[i] = f2bf(in[i]);
}

// ------------------------------------------------------------- WMMA GEMM ---
// C[M,N](f32) = A[M,K](bf16,row-major) * B[K,N](bf16,row-major) + bias[N]
// 256 threads = 8 waves (2x4). Block tile 128x128, K-step 64. A tile staged
// by the TDM (hardware pad -> LDT_S stride); B staged manually (transpose).
#define BM 128
#define BN 128
#define BK 64
#define LDT_S 72   // padded LDS stride: 128B data + 16B pad = 144B per row

template <int N, int K>
__global__ __launch_bounds__(256)
void gemm_bf16(const unsigned short* __restrict__ A,
               const unsigned short* __restrict__ B,
               const float* __restrict__ bias,
               float* __restrict__ C) {
  __shared__ unsigned short As[BM * LDT_S];  // [m][k] (TDM-padded)
  __shared__ unsigned short Bs[BN * LDT_S];  // [n][k] (transposed)
  const int tid  = threadIdx.x;
  const int lane = tid & 31;
  const int wave = tid >> 5;
  const int g    = lane >> 4;   // lane half-group
  const int ln   = lane & 15;
  const int wm   = wave >> 2;   // 0..1 -> 64 rows each
  const int wn   = wave & 3;    // 0..3 -> 32 cols each
  const int mBase = blockIdx.y * BM;
  const int nBase = blockIdx.x * BN;
  const unsigned asLds = (unsigned)(unsigned long long)(const void*)As;

  v8f z = {};
  v8f acc[4][2];
  for (int mt = 0; mt < 4; ++mt)
    for (int nt = 0; nt < 2; ++nt) acc[mt][nt] = z;

  for (int kt = 0; kt < K; kt += BK) {
    // Stage A tile (128 x 64 bf16 = 128 rows x 128B) via Tensor Data Mover.
    // Row = 128B = 32 dwords -> pad_interval code 4; +16B pad -> 144B stride.
    if (wave == 0) {
      tdm_load_2d(asLds, A + (size_t)mBase * K + kt, G1W0_PAD(4),
                  /*tdim0=*/K / 4, /*tdim1=*/BM,
                  /*tile0=*/BK / 4, /*tile1=*/BM, /*stride0=*/K / 4);
    }
    // Stage B tile (64x128) transposed into Bs[n][k].
#pragma unroll
    for (int i = 0; i < 4; ++i) {
      int c  = tid + i * 256;
      int kr = c >> 4;
      int nc = (c & 15) * 8;
      uint4 v = *(const uint4*)(B + (size_t)(kt + kr) * N + nBase + nc);
      const unsigned short* s = (const unsigned short*)&v;
#pragma unroll
      for (int j = 0; j < 8; ++j) Bs[(nc + j) * LDT_S + kr] = s[j];
    }
    // Prefetch next B tile into cache while this stage computes.
    if (kt + BK < K)
      __builtin_prefetch(B + (size_t)(kt + BK + (tid >> 4)) * N + nBase + (tid & 15) * 8, 0, 1);
    if (wave == 0) __builtin_amdgcn_s_wait_tensorcnt(0);
    __syncthreads();

#pragma unroll
    for (int kc = 0; kc < 2; ++kc) {
      // A fragment (16x32 bf16): lane ln = row m; elems 0..7 = K[g*8..+7],
      // elems 8..15 = K[16+g*8..+7]  (ISA 7.12.2 16-bit A layout).
      v16bf af[4], bfrag[2];
#pragma unroll
      for (int mt = 0; mt < 4; ++mt) {
        int m = wm * 64 + mt * 16 + ln;
        const unsigned short* p = &As[m * LDT_S + kc * 32];
        af[mt] = ld_frag(p + g * 8, p + 16 + g * 8);
      }
      // B fragment (32x16 bf16): lane ln = column n; elems 0..7 = K[g*8..+7], 8..15 = K[16+g*8..+7].
#pragma unroll
      for (int nt = 0; nt < 2; ++nt) {
        int n = wn * 32 + nt * 16 + ln;
        const unsigned short* p = &Bs[n * LDT_S + kc * 32 + g * 8];
        bfrag[nt] = ld_frag(p, p + 16);
      }
#pragma unroll
      for (int mt = 0; mt < 4; ++mt)
#pragma unroll
        for (int nt = 0; nt < 2; ++nt)
          acc[mt][nt] = __builtin_amdgcn_wmma_f32_16x16x32_bf16(
              false, af[mt], false, bfrag[nt], (short)0, acc[mt][nt], false, false);
    }
    __syncthreads();
  }

  // Epilogue: C layout VGPR r, lane: M = r + 8*g, N = ln.
#pragma unroll
  for (int nt = 0; nt < 2; ++nt) {
    int n = nBase + wn * 32 + nt * 16 + ln;
    float bv = bias[n];
#pragma unroll
    for (int mt = 0; mt < 4; ++mt) {
      int m0 = mBase + wm * 64 + mt * 16 + g * 8;
#pragma unroll
      for (int r = 0; r < 8; ++r)
        C[(size_t)(m0 + r) * N + n] = acc[mt][nt][r] + bv;
    }
  }
}

// -------------------------------------------- RoPE + head-major relayout ---
// qkv f32 [B*L, 3E] -> Q/K (rotated) and V as bf16 [B,H,L,D].
__global__ void rope_split(const float* __restrict__ qkv,
                           unsigned short* __restrict__ Qb,
                           unsigned short* __restrict__ Kb,
                           unsigned short* __restrict__ Vb) {
  const int total = BATCH * SEQL * NH * (HD / 2);   // 4,194,304
  int i = blockIdx.x * blockDim.x + threadIdx.x;
  int stride = gridDim.x * blockDim.x;
  for (; i < total; i += stride) {
    int j = i & 63;            // pair index within head dim
    int t = i >> 6;
    int h = t & (NH - 1);
    t >>= 4;
    int l = t & (SEQL - 1);
    int b = t >> 11;
    size_t row = (size_t)(b * SEQL + l) * (3 * EMB);
    int col = h * HD + 2 * j;
    float q0 = qkv[row + col],           q1 = qkv[row + col + 1];
    float k0 = qkv[row + EMB + col],     k1 = qkv[row + EMB + col + 1];
    float v0 = qkv[row + 2 * EMB + col], v1 = qkv[row + 2 * EMB + col + 1];
    float freq = __expf(-((float)j / 64.0f) * __logf(10000.0f));
    float s, c;
    __sincosf((float)l * freq, &s, &c);
    size_t o = ((size_t)(b * NH + h) * SEQL + l) * HD + 2 * j;
    Qb[o]     = f2bf(q0 * c - q1 * s);
    Qb[o + 1] = f2bf(q1 * c + q0 * s);
    Kb[o]     = f2bf(k0 * c - k1 * s);
    Kb[o + 1] = f2bf(k1 * c + k0 * s);
    Vb[o]     = f2bf(v0);
    Vb[o + 1] = f2bf(v1);
  }
}

// -------------------------------------------------------- flash attention ---
// One block = 4 waves; wave w owns 16 queries x D=128. K tile staged by TDM
// (hardware pad -> 136-short stride); V staged manually (transposed).
__global__ __launch_bounds__(128)
void flash_attn(const unsigned short* __restrict__ Qb,
                const unsigned short* __restrict__ Kb,
                const unsigned short* __restrict__ Vb,
                unsigned short* __restrict__ ctx) {
  __shared__ unsigned short Kt[32 * 136];    // [key][d], TDM-padded
  __shared__ unsigned short Vt[128 * 40];    // [d][key], padded
  __shared__ unsigned short Pb[4 * 16 * 40]; // per-wave P staging [m][k]
  const int tid  = threadIdx.x;
  const int lane = tid & 31;
  const int w    = tid >> 5;
  const int g    = lane >> 4;
  const int ln   = lane & 15;
  const int bh   = blockIdx.x;
  const int b    = bh >> 4, h = bh & (NH - 1);
  const int qg   = blockIdx.y;
  const int qTile = qg * 64 + w * 16;
  const float scale = 0.08838834764831845f;  // 1/sqrt(128)
  const unsigned ktLds = (unsigned)(unsigned long long)(const void*)Kt;

  const unsigned short* Qp = Qb + (size_t)bh * SEQL * HD;
  const unsigned short* Kp = Kb + (size_t)bh * SEQL * HD;
  const unsigned short* Vp = Vb + (size_t)bh * SEQL * HD;

  // Q fragments (A-layout), held for the whole kernel.
  v16bf qf[4];
  {
    const unsigned short* qr = Qp + (size_t)(qTile + ln) * HD;
#pragma unroll
    for (int c2 = 0; c2 < 4; ++c2)
      qf[c2] = ld_frag(qr + c2 * 32 + g * 8, qr + c2 * 32 + 16 + g * 8);
  }

  v8f z = {};
  v8f oacc[8];
#pragma unroll
  for (int d = 0; d < 8; ++d) oacc[d] = z;
  float mrow[8], lrow[8];
#pragma unroll
  for (int r = 0; r < 8; ++r) { mrow[r] = -3.0e38f; lrow[r] = 0.0f; }

  const int nKB = 2 * (qg + 1);              // causal: keys up to qg*64+63
  for (int kb = 0; kb < nKB; ++kb) {
    const int kBase = kb * 32;
    // K tile (32 rows x 256B) via TDM: row = 64 dwords -> pad_interval code 5,
    // +16B pad -> 272B stride = Kt[136].
    if (w == 0) {
      tdm_load_2d(ktLds, Kp + (size_t)kBase * HD, G1W0_PAD(5),
                  /*tdim0=*/HD / 4, /*tdim1=*/SEQL,
                  /*tile0=*/HD / 4, /*tile1=*/32, /*stride0=*/HD / 4);
    }
    // Stage V transposed: Vt[d][key].
#pragma unroll
    for (int i = 0; i < 4; ++i) {
      int c   = tid + i * 128;
      int row = c >> 4;
      int dc  = (c & 15) * 8;
      uint4 vv = *(const uint4*)(Vp + (size_t)(kBase + row) * HD + dc);
      const unsigned short* s = (const unsigned short*)&vv;
#pragma unroll
      for (int j = 0; j < 8; ++j) Vt[(dc + j) * 40 + row] = s[j];
    }
    if (w == 0) __builtin_amdgcn_s_wait_tensorcnt(0);
    __syncthreads();

    // S = Q * K^T : B-frag column = key (ln), contraction d in A-table order.
    v8f sacc[2];
    sacc[0] = z; sacc[1] = z;
#pragma unroll
    for (int nt = 0; nt < 2; ++nt)
#pragma unroll
      for (int c2 = 0; c2 < 4; ++c2) {
        const unsigned short* kp = &Kt[(nt * 16 + ln) * 136 + c2 * 32 + g * 8];
        v16bf bfr = ld_frag(kp, kp + 16);
        sacc[nt] = __builtin_amdgcn_wmma_f32_16x16x32_bf16(
            false, qf[c2], false, bfr, (short)0, sacc[nt], false, false);
      }

    // Scale, causal mask, online softmax (per-row stats live in VGPRs).
    float alpha[8];
    unsigned short* Pw = &Pb[w * 16 * 40];
#pragma unroll
    for (int r = 0; r < 8; ++r) {
      int q = qTile + r + 8 * g;
      float s0 = sacc[0][r] * scale, s1 = sacc[1][r] * scale;
      if (kBase + ln      > q) s0 = -1.0e30f;
      if (kBase + 16 + ln > q) s1 = -1.0e30f;
      float rm = fmaxf(s0, s1);
#pragma unroll
      for (int msk = 1; msk < 16; msk <<= 1)
        rm = fmaxf(rm, __shfl_xor(rm, msk, 32));
      float mnew = fmaxf(mrow[r], rm);
      alpha[r] = __expf(mrow[r] - mnew);
      float p0 = __expf(s0 - mnew), p1 = __expf(s1 - mnew);
      float rs = p0 + p1;
#pragma unroll
      for (int msk = 1; msk < 16; msk <<= 1)
        rs += __shfl_xor(rs, msk, 32);
      lrow[r] = alpha[r] * lrow[r] + rs;
      mrow[r] = mnew;
      // C-layout -> A-layout transpose through LDS (bf16).
      Pw[(r + 8 * g) * 40 + ln]      = f2bf(p0);
      Pw[(r + 8 * g) * 40 + 16 + ln] = f2bf(p1);
    }
    // Same-wave DS ops are in-order; wait for staging stores, block reorder.
    asm volatile("s_wait_dscnt 0" ::: "memory");
    v16bf pf = ld_frag(&Pw[ln * 40 + g * 8], &Pw[ln * 40 + 16 + g * 8]);

    // O = diag(alpha) * O + P * V
#pragma unroll
    for (int dt = 0; dt < 8; ++dt) {
#pragma unroll
      for (int r = 0; r < 8; ++r) oacc[dt][r] = oacc[dt][r] * alpha[r];
      const unsigned short* vp = &Vt[(dt * 16 + ln) * 40 + g * 8];
      v16bf vf = ld_frag(vp, vp + 16);
      oacc[dt] = __builtin_amdgcn_wmma_f32_16x16x32_bf16(
          false, pf, false, vf, (short)0, oacc[dt], false, false);
    }
    __syncthreads();
  }

  // Normalize and emit ctx (bf16, [B, L, E] so the out-GEMM reads row-major).
#pragma unroll
  for (int r = 0; r < 8; ++r) {
    float inv = 1.0f / lrow[r];
    int q = qTile + r + 8 * g;
    size_t base = ((size_t)(b * SEQL + q)) * EMB + h * HD;
#pragma unroll
    for (int dt = 0; dt < 8; ++dt)
      ctx[base + dt * 16 + ln] = f2bf(oacc[dt][r] * inv);
  }
}

// ---------------------------------------------------------------- launch ---
extern "C" void kernel_launch(void* const* d_in, const int* in_sizes, int n_in,
                              void* d_out, int out_size, void* d_ws, size_t ws_size,
                              hipStream_t stream) {
  (void)in_sizes; (void)n_in; (void)out_size; (void)ws_size;
  const int threeE = 3 * EMB;
  const int Mrows  = BATCH * SEQL;          // 4096

  const float* x    = (const float*)d_in[0];
  const float* Wqkv = (const float*)d_in[1];
  const float* bqkv = (const float*)d_in[2];
  const float* Wout = (const float*)d_in[3];
  const float* bout = (const float*)d_in[4];
  float* out = (float*)d_out;

  char* ws = (char*)d_ws;
  size_t off = 0;
  auto alloc = [&](size_t bytes) -> void* {
    void* p = ws + off;
    off = (off + bytes + 255) & ~(size_t)255;
    return p;
  };
  unsigned short* xbf    = (unsigned short*)alloc((size_t)Mrows * EMB * 2);
  unsigned short* wqkvbf = (unsigned short*)alloc((size_t)EMB * threeE * 2);
  unsigned short* woutbf = (unsigned short*)alloc((size_t)EMB * EMB * 2);
  float*          qkv    = (float*)alloc((size_t)Mrows * threeE * 4);
  unsigned short* Qbf    = (unsigned short*)alloc((size_t)BATCH * NH * SEQL * HD * 2);
  unsigned short* Kbf    = (unsigned short*)alloc((size_t)BATCH * NH * SEQL * HD * 2);
  unsigned short* Vbf    = (unsigned short*)alloc((size_t)BATCH * NH * SEQL * HD * 2);
  unsigned short* ctxbf  = (unsigned short*)alloc((size_t)Mrows * EMB * 2);

  // 1) fp32 -> bf16 casts
  cast_f32_bf16<<<2048, 256, 0, stream>>>(x,    xbf,    Mrows * EMB);
  cast_f32_bf16<<<2048, 256, 0, stream>>>(Wqkv, wqkvbf, EMB * threeE);
  cast_f32_bf16<<<2048, 256, 0, stream>>>(Wout, woutbf, EMB * EMB);

  // 2) qkv = x @ Wqkv + bqkv           [4096 x 6144] f32
  gemm_bf16<3 * EMB, EMB><<<dim3(threeE / BN, Mrows / BM), 256, 0, stream>>>(
      xbf, wqkvbf, bqkv, qkv);

  // 3) RoPE(q,k) + relayout to [B,H,L,D] bf16
  rope_split<<<8192, 256, 0, stream>>>(qkv, Qbf, Kbf, Vbf);

  // 4) causal flash attention -> ctx bf16 [B, L, E]
  flash_attn<<<dim3(BATCH * NH, SEQL / 64), 128, 0, stream>>>(
      Qbf, Kbf, Vbf, ctxbf);

  // 5) out = ctx @ Wout + bout          [4096 x 2048] f32
  gemm_bf16<EMB, EMB><<<dim3(EMB / BN, Mrows / BM), 256, 0, stream>>>(
      ctxbf, woutbf, bout, out);
}
